// lstm_56238301774012
// MI455X (gfx1250) — matrix operand order repaired and checked by reading, not compiled, not run on hardware
//
#include <hip/hip_runtime.h>
#include <hip/hip_bf16.h>
#include <math.h>

// ---------------------------------------------------------------------------
// Problem constants (from the reference)
// ---------------------------------------------------------------------------
#define BATCH 8192
#define HID   1024
#define KXH   2048          // concat [x|h] inner dim
#define D1DIM 512
#define D2VAL 341
#define D2PAD 352           // 341 padded to 22*16
#define ALPH  512

// ---------------------------------------------------------------------------
// WMMA types (CDNA5 wave32): v16bf A/B fragments, v8f C/D accumulator
// ---------------------------------------------------------------------------
typedef __attribute__((ext_vector_type(16))) __bf16       v16bf;
typedef __attribute__((ext_vector_type(8)))  float        v8f;
typedef __attribute__((ext_vector_type(4)))  unsigned int u32x4;

union Frag { v16bf v; u32x4 q[2]; };

__device__ __forceinline__ unsigned short f2bf(float x) {
  union { float f; unsigned int u; } t; t.f = x;
  unsigned int u = t.u;
  u += 0x7fffu + ((u >> 16) & 1u);          // round-to-nearest-even
  return (unsigned short)(u >> 16);
}
__device__ __forceinline__ float sigm(float x) { return 1.0f / (1.0f + __expf(-x)); }

#define WMMA_BF16(A, B, C) \
  __builtin_amdgcn_wmma_f32_16x16x32_bf16(false, (A), false, (B), (short)0, (C), false, false)

// ---------------------------------------------------------------------------
// Conversion / packing kernels
// ---------------------------------------------------------------------------
__global__ void build_xh(const float* __restrict__ x, const float* __restrict__ h,
                         unsigned short* __restrict__ xh) {
  long long idx = (long long)blockIdx.x * blockDim.x + threadIdx.x;
  const long long total = (long long)BATCH * KXH;
  if (idx >= total) return;
  int b = (int)(idx >> 11);
  int k = (int)(idx & 2047);
  float v = (k < HID) ? x[(size_t)b * HID + k] : h[(size_t)b * HID + (k - HID)];
  xh[idx] = f2bf(v);
}

__global__ void build_wcat(const float* __restrict__ Wfi, const float* __restrict__ Wfh,
                           const float* __restrict__ Wii, const float* __restrict__ Wih,
                           const float* __restrict__ Wsi, const float* __restrict__ Wsh,
                           const float* __restrict__ Woi, const float* __restrict__ Woh,
                           unsigned short* __restrict__ Wcat) {
  long long idx = (long long)blockIdx.x * blockDim.x + threadIdx.x;
  const long long total = (long long)4 * HID * KXH;
  if (idx >= total) return;
  int r = (int)(idx >> 11);     // 0..4095 : gate*1024 + j
  int k = (int)(idx & 2047);    // 0..2047 : [x-part | h-part]
  int g = r >> 10;
  int j = r & 1023;
  const float* Wi = (g == 0) ? Wfi : (g == 1) ? Wii : (g == 2) ? Wsi : Woi;
  const float* Wh = (g == 0) ? Wfh : (g == 1) ? Wih : (g == 2) ? Wsh : Woh;
  float v = (k < HID) ? Wi[(size_t)j * HID + k] : Wh[(size_t)j * HID + (k - HID)];
  Wcat[idx] = f2bf(v);
}

__global__ void cvt_bf(const float* __restrict__ src, unsigned short* __restrict__ dst, int n) {
  int i = blockIdx.x * blockDim.x + threadIdx.x;
  if (i < n) dst[i] = f2bf(src[i]);
}

__global__ void cvt_pad(const float* __restrict__ src, int inR, int inC,
                        unsigned short* __restrict__ dst, int outR, int outC) {
  int i = blockIdx.x * blockDim.x + threadIdx.x;
  if (i >= outR * outC) return;
  int r = i / outC, c = i % outC;
  float v = (r < inR && c < inC) ? src[(size_t)r * inC + c] : 0.0f;
  dst[i] = f2bf(v);
}

// ---------------------------------------------------------------------------
// Fused gate GEMM + LSTM cell update, register-blocked.
// Block = 128 threads (4 waves). Wave w owns batch rows [by*128+w*32, +32)
// (two 16-row M-tiles), block owns hidden cols [bx*16, +16).
// 8 accumulators (4 gates x 2 M-tiles); per K-step: 2 A frags + 4 B frags
// feed 8 WMMAs (1.5 b128 loads per WMMA, each B frag reused twice).
// ---------------------------------------------------------------------------
__global__ __launch_bounds__(128) void gates_kernel(
    const unsigned short* __restrict__ xh,     // [B x 2048] bf16
    const unsigned short* __restrict__ Wcat,   // [4096 x 2048] bf16
    const float* __restrict__ c,
    const float* __restrict__ bfi, const float* __restrict__ bfh,
    const float* __restrict__ bii, const float* __restrict__ bih,
    const float* __restrict__ bsi, const float* __restrict__ bsh,
    const float* __restrict__ boi, const float* __restrict__ boh,
    float* __restrict__ hout,                  // [B x 1024] f32 (d_out part 1)
    unsigned short* __restrict__ hbf)          // [B x 1024] bf16 (scratch)
{
  const int lane = threadIdx.x & 31;
  const int wv   = threadIdx.x >> 5;
  const int nl   = lane & 15;
  const int hi   = lane >> 4;
  const int j0   = blockIdx.x * 16;
  const int rowBase = blockIdx.y * 128 + wv * 32;

  // A fragments: lane holds row (rowBase + mt*16 + nl), K segs hi*8+{0..7,16..23}
  const unsigned short* pA0 = xh + (size_t)(rowBase + nl)      * KXH + hi * 8;
  const unsigned short* pA1 = xh + (size_t)(rowBase + 16 + nl) * KXH + hi * 8;
  // B fragments: lane holds W row (gate*1024 + j0 + nl), K segs hi*8+{0..7,16..23}
  const unsigned short* pB[4];
#pragma unroll
  for (int g = 0; g < 4; ++g)
    pB[g] = Wcat + (size_t)(g * HID + j0 + nl) * KXH + hi * 8;

  v8f acc[8] = {};                // acc[g*2 + mt]

  for (int kk = 0; kk < KXH; kk += 32) {
    __builtin_prefetch(pA0 + kk + 128, 0, 1);   // global_prefetch_b8
    Frag a0, a1, b;
    a0.q[0] = *(const u32x4*)(pA0 + kk);
    a0.q[1] = *(const u32x4*)(pA0 + kk + 16);
    a1.q[0] = *(const u32x4*)(pA1 + kk);
    a1.q[1] = *(const u32x4*)(pA1 + kk + 16);
#pragma unroll
    for (int g = 0; g < 4; ++g) {
      b.q[0] = *(const u32x4*)(pB[g] + kk);
      b.q[1] = *(const u32x4*)(pB[g] + kk + 16);
      acc[g * 2 + 0] = WMMA_BF16(a0.v, b.v, acc[g * 2 + 0]);
      acc[g * 2 + 1] = WMMA_BF16(a1.v, b.v, acc[g * 2 + 1]);
    }
  }

  // C/D layout: VGPR v, lane -> (M = v + 8*hi, N = nl)
  const int n = j0 + nl;
  const float bf_ = bfi[n] + bfh[n];
  const float bi_ = bii[n] + bih[n];
  const float bs_ = bsi[n] + bsh[n];
  const float bo_ = boi[n] + boh[n];
#pragma unroll
  for (int mt = 0; mt < 2; ++mt) {
#pragma unroll
    for (int v = 0; v < 8; ++v) {
      const int m = rowBase + mt * 16 + v + hi * 8;
      const float fg = sigm(acc[0 * 2 + mt][v] + bf_);
      const float ig = sigm(acc[1 * 2 + mt][v] + bi_);
      const float sg = tanhf(acc[2 * 2 + mt][v] + bs_);
      const float og = sigm(acc[3 * 2 + mt][v] + bo_);
      const float cn = c[(size_t)m * HID + n] * fg + ig * sg;
      const float hn = og * tanhf(cn);
      hout[(size_t)m * HID + n] = hn;
      hbf[(size_t)m * HID + n]  = f2bf(hn);
    }
  }
}

// ---------------------------------------------------------------------------
// Generic bf16 GEMM + tanh epilogue:  out = tanh(A[MxK] @ W[NxK]^T + bias)
// Register-blocked M4 x N2: wave owns 64 rows x 32 cols, 8 accumulators.
// Per K-step: 4 A frags + 2 B frags feed 8 WMMAs (1.5 b128 loads / WMMA).
// ---------------------------------------------------------------------------
__global__ __launch_bounds__(128) void gemm_tanh_bf16(
    const unsigned short* __restrict__ A, int lda,
    const unsigned short* __restrict__ W, int ldw,
    const float* __restrict__ bias, int Nvalid,
    unsigned short* __restrict__ out, int ldo, int K)
{
  const int lane = threadIdx.x & 31;
  const int wv   = threadIdx.x >> 5;
  const int nl   = lane & 15;
  const int hi   = lane >> 4;
  const int j0   = blockIdx.x * 32;
  const int rowBase = blockIdx.y * 256 + wv * 64;

  const unsigned short* pA[4];
#pragma unroll
  for (int mt = 0; mt < 4; ++mt)
    pA[mt] = A + (size_t)(rowBase + mt * 16 + nl) * lda + hi * 8;
  const unsigned short* pB[2];
#pragma unroll
  for (int nt = 0; nt < 2; ++nt)
    pB[nt] = W + (size_t)(j0 + nt * 16 + nl) * ldw + hi * 8;

  v8f acc[8] = {};                // acc[nt*4 + mt]

  for (int kk = 0; kk < K; kk += 32) {
    Frag a[4], b;
#pragma unroll
    for (int mt = 0; mt < 4; ++mt) {
      a[mt].q[0] = *(const u32x4*)(pA[mt] + kk);
      a[mt].q[1] = *(const u32x4*)(pA[mt] + kk + 16);
    }
#pragma unroll
    for (int nt = 0; nt < 2; ++nt) {
      b.q[0] = *(const u32x4*)(pB[nt] + kk);
      b.q[1] = *(const u32x4*)(pB[nt] + kk + 16);
#pragma unroll
      for (int mt = 0; mt < 4; ++mt)
        acc[nt * 4 + mt] = WMMA_BF16(a[mt].v, b.v, acc[nt * 4 + mt]);
    }
  }

#pragma unroll
  for (int nt = 0; nt < 2; ++nt) {
    const int n = j0 + nt * 16 + nl;
    const float bb = (n < Nvalid) ? bias[n] : 0.0f;
#pragma unroll
    for (int mt = 0; mt < 4; ++mt) {
#pragma unroll
      for (int v = 0; v < 8; ++v) {
        const int m = rowBase + mt * 16 + v + hi * 8;
        out[(size_t)m * ldo + n] = f2bf(tanhf(acc[nt * 4 + mt][v] + bb));
      }
    }
  }
}

// ---------------------------------------------------------------------------
// Final layer: logits = d2 @ W_d3^T + b_d3 into LDS, then row softmax.
// Block = 128 threads (4 waves) handles 16 batch rows x 512 outputs.
// N-blocked by 2 so each A fragment feeds two WMMAs.
// ---------------------------------------------------------------------------
__global__ __launch_bounds__(128) void logits_softmax(
    const unsigned short* __restrict__ A,    // [B x 352] bf16
    const unsigned short* __restrict__ W,    // [512 x 352] bf16
    const float* __restrict__ bias,          // [512]
    float* __restrict__ out)                 // [B x 512] f32
{
  __shared__ float sm[16 * ALPH];            // 32 KB of the 320 KB WGP LDS
  const int lane = threadIdx.x & 31;
  const int wv   = threadIdx.x >> 5;
  const int nl   = lane & 15;
  const int hi   = lane >> 4;
  const int rowBase = blockIdx.x * 16;

  const unsigned short* pA = A + (size_t)(rowBase + nl) * D2PAD + hi * 8;

  for (int t = 0; t < 4; ++t) {              // 4 waves x 4 pairs = 32 N-tiles
    const int nt0 = wv * 8 + t * 2;
    const unsigned short* pB0 = W + (size_t)((nt0 + 0) * 16 + nl) * D2PAD + hi * 8;
    const unsigned short* pB1 = W + (size_t)((nt0 + 1) * 16 + nl) * D2PAD + hi * 8;
    v8f acc0 = {}, acc1 = {};
    for (int kk = 0; kk < D2PAD; kk += 32) { // 352/32 = 11 iterations
      Frag a, b0, b1;
      a.q[0]  = *(const u32x4*)(pA + kk);
      a.q[1]  = *(const u32x4*)(pA + kk + 16);
      b0.q[0] = *(const u32x4*)(pB0 + kk);
      b0.q[1] = *(const u32x4*)(pB0 + kk + 16);
      b1.q[0] = *(const u32x4*)(pB1 + kk);
      b1.q[1] = *(const u32x4*)(pB1 + kk + 16);
      acc0 = WMMA_BF16(a.v, b0.v, acc0);
      acc1 = WMMA_BF16(a.v, b1.v, acc1);
    }
    const int n0 = (nt0 + 0) * 16 + nl;
    const int n1 = (nt0 + 1) * 16 + nl;
    const float bb0 = bias[n0];
    const float bb1 = bias[n1];
#pragma unroll
    for (int v = 0; v < 8; ++v) {
      sm[(v + hi * 8) * ALPH + n0] = acc0[v] + bb0;
      sm[(v + hi * 8) * ALPH + n1] = acc1[v] + bb1;
    }
  }
  __syncthreads();

  // Softmax: 8 threads per row, cross-lane reduce within aligned groups of 8.
  const int r = threadIdx.x >> 3;
  const int s = threadIdx.x & 7;
  float* row = sm + r * ALPH;
  float mx = -1e30f;
  for (int cidx = s; cidx < ALPH; cidx += 8) mx = fmaxf(mx, row[cidx]);
  mx = fmaxf(mx, __shfl_xor(mx, 1));
  mx = fmaxf(mx, __shfl_xor(mx, 2));
  mx = fmaxf(mx, __shfl_xor(mx, 4));
  float sum = 0.0f;
  for (int cidx = s; cidx < ALPH; cidx += 8) {
    float e = __expf(row[cidx] - mx);
    row[cidx] = e;
    sum += e;
  }
  sum += __shfl_xor(sum, 1);
  sum += __shfl_xor(sum, 2);
  sum += __shfl_xor(sum, 4);
  const float inv = 1.0f / sum;
  float* orow = out + (size_t)(rowBase + r) * ALPH;
  for (int cidx = s; cidx < ALPH; cidx += 8) orow[cidx] = row[cidx] * inv;
}

// ---------------------------------------------------------------------------
// Host launcher
// ---------------------------------------------------------------------------
extern "C" void kernel_launch(void* const* d_in, const int* in_sizes, int n_in,
                              void* d_out, int out_size, void* d_ws, size_t ws_size,
                              hipStream_t stream) {
  const float* x   = (const float*)d_in[0];
  const float* h   = (const float*)d_in[1];
  const float* c   = (const float*)d_in[2];
  const float* Wfi = (const float*)d_in[3],  *bfi = (const float*)d_in[4];
  const float* Wfh = (const float*)d_in[5],  *bfh = (const float*)d_in[6];
  const float* Wii = (const float*)d_in[7],  *bii = (const float*)d_in[8];
  const float* Wih = (const float*)d_in[9],  *bih = (const float*)d_in[10];
  const float* Wsi = (const float*)d_in[11], *bsi = (const float*)d_in[12];
  const float* Wsh = (const float*)d_in[13], *bsh = (const float*)d_in[14];
  const float* Woi = (const float*)d_in[15], *boi = (const float*)d_in[16];
  const float* Woh = (const float*)d_in[17], *boh = (const float*)d_in[18];
  const float* Wd1 = (const float*)d_in[19], *bd1 = (const float*)d_in[20];
  const float* Wd2 = (const float*)d_in[21], *bd2 = (const float*)d_in[22];
  const float* Wd3 = (const float*)d_in[23], *bd3 = (const float*)d_in[24];

  char* ws = (char*)d_ws;
  size_t off = 0;
  auto carve = [&](size_t bytes) -> void* {
    void* p = ws + off;
    off += (bytes + 255) & ~(size_t)255;
    return p;
  };
  unsigned short* xh    = (unsigned short*)carve((size_t)BATCH * KXH * 2);     // 32 MB
  unsigned short* wcat  = (unsigned short*)carve((size_t)4 * HID * KXH * 2);   // 16 MB
  unsigned short* hbf   = (unsigned short*)carve((size_t)BATCH * HID * 2);     // 16 MB
  unsigned short* wd1b  = (unsigned short*)carve((size_t)D1DIM * HID * 2);     //  1 MB
  unsigned short* d1b   = (unsigned short*)carve((size_t)BATCH * D1DIM * 2);   //  8 MB
  unsigned short* wd2b  = (unsigned short*)carve((size_t)D2PAD * D1DIM * 2);
  unsigned short* d2b   = (unsigned short*)carve((size_t)BATCH * D2PAD * 2);   //  6 MB
  unsigned short* wd3b  = (unsigned short*)carve((size_t)ALPH * D2PAD * 2);

  float* hout = (float*)d_out;                                  // [B x 1024]
  float* dout = (float*)d_out + (size_t)BATCH * HID;            // [B x 512]

  // 1) Pack inputs / weights to bf16
  {
    long long n = (long long)BATCH * KXH;
    build_xh<<<(unsigned)((n + 255) / 256), 256, 0, stream>>>(x, h, xh);
  }
  {
    long long n = (long long)4 * HID * KXH;
    build_wcat<<<(unsigned)((n + 255) / 256), 256, 0, stream>>>(
        Wfi, Wfh, Wii, Wih, Wsi, Wsh, Woi, Woh, wcat);
  }
  {
    int n = D1DIM * HID;
    cvt_bf<<<(n + 255) / 256, 256, 0, stream>>>(Wd1, wd1b, n);
  }
  {
    int n = D2PAD * D1DIM;
    cvt_pad<<<(n + 255) / 256, 256, 0, stream>>>(Wd2, D2VAL, D1DIM, wd2b, D2PAD, D1DIM);
  }
  {
    int n = ALPH * D2PAD;
    cvt_pad<<<(n + 255) / 256, 256, 0, stream>>>(Wd3, ALPH, D2VAL, wd3b, ALPH, D2PAD);
  }

  // 2) Fused 4-gate GEMM + LSTM cell update -> h_new
  gates_kernel<<<dim3(HID / 16, BATCH / 128), 128, 0, stream>>>(
      xh, wcat, c, bfi, bfh, bii, bih, bsi, bsh, boi, boh, hout, hbf);

  // 3) Decoder layers 1 & 2 (tanh epilogues)
  gemm_tanh_bf16<<<dim3(D1DIM / 32, BATCH / 256), 128, 0, stream>>>(
      hbf, HID, wd1b, HID, bd1, D1DIM, d1b, D1DIM, HID);
  gemm_tanh_bf16<<<dim3(D2PAD / 32, BATCH / 256), 128, 0, stream>>>(
      d1b, D1DIM, wd2b, D1DIM, bd2, D2VAL, d2b, D2PAD, D1DIM);

  // 4) Final projection + softmax
  logits_softmax<<<BATCH / 16, 128, 0, stream>>>(d2b, wd3b, bd3, dout);
}
